// MultiHeadAlibiSelfAttentionLayer_4544075399526
// MI455X (gfx1250) — matrix operand order repaired and checked by reading, not compiled, not run on hardware
//
#include <hip/hip_runtime.h>

// ---------------------------------------------------------------------------
// MultiHeadAlibiSelfAttentionLayer for MI455X (gfx1250, wave32, WMMA)
// B=2, S=2048, D=1024, H=16, HD=64
// Pipeline: 3x GEMM (x@W + b -> bf16 [B,H,S,64])  ->  flash attention (bf16
// WMMA, f32 accum, ALiBi + key-padding mask, double-buffered async K-tile
// DMA to LDS)  ->  GEMM (attn@Wo + bo, f32).  GEMMs are LDS double-buffered
// with sched barriers pinning the load/compute pipeline shape.
// ---------------------------------------------------------------------------

#define BB  2
#define SS  2048
#define DD  1024
#define HH  16
#define HDD 64

#define SCHED_FENCE() __builtin_amdgcn_sched_barrier(0)

typedef __attribute__((ext_vector_type(16))) __bf16        v16bf;
typedef __attribute__((ext_vector_type(8)))  float         v8f;
typedef __attribute__((ext_vector_type(4))) unsigned int   u32x4;
typedef __attribute__((ext_vector_type(4)))  float         f32x4;

union Frag { v16bf v; u32x4 q[2]; };

__device__ __forceinline__ unsigned short f2bfu(float f) {
  unsigned int u = __builtin_bit_cast(unsigned int, f);
  u += 0x7FFFu + ((u >> 16) & 1u);   // round-to-nearest-even
  return (unsigned short)(u >> 16);
}

__device__ __forceinline__ v8f wmma_bf16(const Frag& a, const Frag& b, v8f c) {
  // (neg_a, A, neg_b, B, c_mod, C, reuse_a, reuse_b)
  return __builtin_amdgcn_wmma_f32_16x16x32_bf16(false, a.v, false, b.v,
                                                 (short)0, c, false, false);
}

// Stage one k-tile (A: 128x32 rows, W: 32x128 transposed to [n][k]) into LDS.
__device__ __forceinline__ void stage_tiles(unsigned short* __restrict__ dA,
                                            unsigned short* __restrict__ dB,
                                            const f32x4* ra, const f32x4* rw,
                                            int arow, int ac4, int wk, int wn4)
{
  for (int i = 0; i < 4; ++i) {
    unsigned p0 = (unsigned)f2bfu(ra[i].x) | ((unsigned)f2bfu(ra[i].y) << 16);
    unsigned p1 = (unsigned)f2bfu(ra[i].z) | ((unsigned)f2bfu(ra[i].w) << 16);
    unsigned* dst = (unsigned*)&dA[(arow + i * 32) * 32 + ac4 * 4];
    dst[0] = p0; dst[1] = p1;
    const int k = wk + i * 8;
    dB[(wn4 * 4 + 0) * 32 + k] = f2bfu(rw[i].x);
    dB[(wn4 * 4 + 1) * 32 + k] = f2bfu(rw[i].y);
    dB[(wn4 * 4 + 2) * 32 + k] = f2bfu(rw[i].z);
    dB[(wn4 * 4 + 3) * 32 + k] = f2bfu(rw[i].w);
  }
}

// ---------------------------------------------------------------------------
// GEMM: out[M=4096, N=1024] = X[M,K=1024] * W[K,N] + bias[N]
// mode 0: out = f32, row-major [M,N]
// mode 1: out = bf16, head-split [B,H,S,HD] (for Q/K/V)
// Tile: 128x128 per workgroup (256 thr = 8 waves), K-step 32, double-buffered.
// ---------------------------------------------------------------------------
__global__ __launch_bounds__(256)
void mha_gemm_bf16wmma(const float* __restrict__ X, const float* __restrict__ W,
                       const float* __restrict__ bias, void* __restrict__ out,
                       int mode)
{
  __shared__ unsigned short lA[2][128 * 32];   // [row][k]  bf16
  __shared__ unsigned short lB[2][128 * 32];   // [n][k]    bf16 (W transposed)

  const int tid  = threadIdx.x;
  const int lane = tid & 31;
  const int wave = tid >> 5;
  const int hsel = (lane >> 4) & 1;
  const int l15  = lane & 15;

  const int row0 = blockIdx.y * 128;
  const int n0   = blockIdx.x * 128;

  const int arow = tid >> 3, ac4 = tid & 7;    // A loader: rows arow+{0,32,64,96}
  const int wk   = tid >> 5, wn4 = tid & 31;   // W loader: k rows wk+{0,8,16,24}

  v8f acc[8];
  for (int t = 0; t < 8; ++t)
    for (int j = 0; j < 8; ++j) acc[t][j] = 0.0f;

  f32x4 ra[4], rw[4];

  // Prologue: k-tile 0 -> buffer 0
  for (int i = 0; i < 4; ++i) {
    ra[i] = *(const f32x4*)&X[(row0 + arow + i * 32) * DD + ac4 * 4];
    rw[i] = *(const f32x4*)&W[(wk + i * 8) * DD + n0 + wn4 * 4];
  }
  stage_tiles(lA[0], lB[0], ra, rw, arow, ac4, wk, wn4);
  __syncthreads();

  int buf = 0;
  for (int kt = 0; kt < DD / 32; ++kt) {
    const bool more = (kt + 1) < (DD / 32);
    if (more) {   // issue next tile's global loads, fenced so they stay early
      const int k0 = (kt + 1) * 32;
      for (int i = 0; i < 4; ++i) {
        ra[i] = *(const f32x4*)&X[(row0 + arow + i * 32) * DD + k0 + ac4 * 4];
        rw[i] = *(const f32x4*)&W[(k0 + wk + i * 8) * DD + n0 + wn4 * 4];
      }
      SCHED_FENCE();
    }

    // A fragment (16x32): lanes 0-15 K{0-7,16-23}, lanes 16-31 K{8-15,24-31}
    Frag a;
    const int ar = wave * 16 + l15;
    a.q[0] = *(const u32x4*)&lA[buf][ar * 32 + hsel * 8];
    a.q[1] = *(const u32x4*)&lA[buf][ar * 32 + 16 + hsel * 8];

    // B fragments in batches of 4; fence keeps all 8 ds_load_b128 ahead of
    // the WMMAs so a single dscnt wait covers the batch.
    for (int g = 0; g < 2; ++g) {
      Frag bfr[4];
      for (int nt = 0; nt < 4; ++nt) {
        const int nr = (g * 4 + nt) * 16 + l15;
        bfr[nt].q[0] = *(const u32x4*)&lB[buf][nr * 32 + hsel * 8];
        bfr[nt].q[1] = *(const u32x4*)&lB[buf][nr * 32 + 16 + hsel * 8];
      }
      SCHED_FENCE();
      for (int nt = 0; nt < 4; ++nt)
        acc[g * 4 + nt] = wmma_bf16(a, bfr[nt], acc[g * 4 + nt]);
      SCHED_FENCE();
    }

    if (more) stage_tiles(lA[buf ^ 1], lB[buf ^ 1], ra, rw, arow, ac4, wk, wn4);
    __syncthreads();
    buf ^= 1;
  }

  // Epilogue: C layout -> VGPR j: M = j + hsel*8, N = l15 (+ tile offsets)
  for (int nt = 0; nt < 8; ++nt) {
    const int N  = n0 + nt * 16 + l15;
    const float bv = bias[N];
    for (int j = 0; j < 8; ++j) {
      const int M   = row0 + wave * 16 + j + hsel * 8;
      const float val = acc[nt][j] + bv;
      if (mode == 0) {
        ((float*)out)[M * DD + N] = val;
      } else {
        const int b = M >> 11, s = M & (SS - 1);
        const int h = N >> 6,  hd = N & (HDD - 1);
        ((unsigned short*)out)[(((b * HH + h) * SS) + s) * HDD + hd] = f2bfu(val);
      }
    }
  }
}

// ---------------------------------------------------------------------------
// Attention tile movers: K via GLOBAL_LOAD_ASYNC_TO_LDS_B128 (ASYNCcnt DMA,
// runs concurrently with WMMA compute), V via registers (needs transpose).
// ---------------------------------------------------------------------------
__device__ __forceinline__ void attn_issue_loads(const unsigned short* __restrict__ K,
                                                 const unsigned short* __restrict__ V,
                                                 unsigned short* __restrict__ dK,
                                                 u32x4* vreg, int off, int kb, int tid)
{
  for (int i = 0; i < 4; ++i) {
    const int idx = tid + i * 128;
    const int r = idx >> 3, c = idx & 7;
    const unsigned ldsK = (unsigned)(size_t)&dK[r * 64 + c * 8];
    const unsigned gof  = (unsigned)((off + (kb + r) * HDD + c * 8) * 2u);
    asm volatile("global_load_async_to_lds_b128 %0, %1, %2 offset:0"
                 :: "v"(ldsK), "v"(gof), "s"(K)
                 : "memory");
    vreg[i] = *(const u32x4*)&V[off + (kb + r) * HDD + c * 8];
  }
}

__device__ __forceinline__ void attn_store_vt(unsigned short* __restrict__ dVt,
                                              const u32x4* vreg, int tid)
{
  for (int i = 0; i < 4; ++i) {
    const int idx = tid + i * 128;
    const int r = idx >> 3, c = idx & 7;
    const unsigned short* vs = (const unsigned short*)&vreg[i];
    for (int j = 0; j < 8; ++j) dVt[(c * 8 + j) * 64 + r] = vs[j];
  }
}

// ---------------------------------------------------------------------------
// Flash attention with ALiBi. Q,K,V: bf16 [B,H,S,64]. out: f32 [B,S,D].
// Block: (S/64, H, B) x 128 threads (4 waves). Wave owns 16 query rows.
// lK/lVt double-buffered: tile kt+1 streams in while tile kt is computed.
// ---------------------------------------------------------------------------
__global__ __launch_bounds__(128)
void mha_attn_flash(const unsigned short* __restrict__ Q,
                    const unsigned short* __restrict__ K,
                    const unsigned short* __restrict__ V,
                    const int* __restrict__ kmask,
                    float* __restrict__ out)
{
  __shared__ unsigned short lK [2][64 * 64];   // [key][hd]
  __shared__ unsigned short lVt[2][64 * 64];   // [hd][key] (transposed)
  __shared__ unsigned short lP [4][16 * 64];   // per-wave P tile

  const int tid  = threadIdx.x;
  const int lane = tid & 31;
  const int wave = tid >> 5;
  const int hsel = (lane >> 4) & 1;
  const int l15  = lane & 15;

  const int qt = blockIdx.x;
  const int h  = blockIdx.y;
  const int b  = blockIdx.z;

  const int off = ((b * HH + h) * SS) * HDD;
  const int q0  = qt * 64 + wave * 16;

  const float slope     = exp2f(-0.5f * (float)(h + 1));  // 2^{-(8/H)*(h+1)}
  const float inv_scale = 0.125f;                         // 1/sqrt(64)

  // Q fragments held in registers across the whole key loop (2 K-steps of 32)
  Frag qf[2];
  {
    const int qr = q0 + l15;
    for (int s = 0; s < 2; ++s) {
      qf[s].q[0] = *(const u32x4*)&Q[off + qr * HDD + s * 32 + hsel * 8];
      qf[s].q[1] = *(const u32x4*)&Q[off + qr * HDD + s * 32 + 16 + hsel * 8];
    }
  }

  float m[8], lsum[8];
  v8f o[4];
  for (int j = 0; j < 8; ++j) { m[j] = -3.0e38f; lsum[j] = 0.0f; }
  for (int t = 0; t < 4; ++t)
    for (int j = 0; j < 8; ++j) o[t][j] = 0.0f;

  u32x4 vreg[4];

  // Prologue: tile 0 -> buffer 0
  attn_issue_loads(K, V, lK[0], vreg, off, 0, tid);
  attn_store_vt(lVt[0], vreg, tid);
  asm volatile("s_wait_asynccnt 0" ::: "memory");
  __syncthreads();

  int buf = 0;
  for (int kt = 0; kt < SS / 64; ++kt) {
    const int kb = kt * 64;
    const bool more = (kt + 1) < (SS / 64);

    if (more) {  // start DMA of tile kt+1 into the alternate buffer now
      attn_issue_loads(K, V, lK[buf ^ 1], vreg, off, kb + 64, tid);
      SCHED_FENCE();
    }

    // S = Q * K^T  (4 score tiles of 16x16, 2 K-steps)
    v8f sc[4];
    for (int t = 0; t < 4; ++t)
      for (int j = 0; j < 8; ++j) sc[t][j] = 0.0f;
    for (int s = 0; s < 2; ++s) {
      Frag bf[4];                                  // B = K^T: n=key, K-dim=hd
      for (int nt = 0; nt < 4; ++nt) {
        const int keyn = nt * 16 + l15;
        bf[nt].q[0] = *(const u32x4*)&lK[buf][keyn * HDD + s * 32 + hsel * 8];
        bf[nt].q[1] = *(const u32x4*)&lK[buf][keyn * HDD + s * 32 + 16 + hsel * 8];
      }
      SCHED_FENCE();
      for (int nt = 0; nt < 4; ++nt)
        sc[nt] = wmma_bf16(qf[s], bf[nt], sc[nt]);
      SCHED_FENCE();
    }

    // ALiBi bias, scale, key-padding mask (per element of the C layout)
    for (int nt = 0; nt < 4; ++nt) {
      const int jg = kb + nt * 16 + l15;
      const bool valid = kmask[b * SS + jg] != 0;
      for (int j = 0; j < 8; ++j) {
        const int ig = q0 + j + hsel * 8;
        const float v = (sc[nt][j] - slope * fabsf((float)(ig - jg))) * inv_scale;
        sc[nt][j] = valid ? v : -1.0e30f;
      }
    }

    // Online softmax (row reductions across the 16-lane half owning the row)
    for (int j = 0; j < 8; ++j) {
      float rm = fmaxf(fmaxf(sc[0][j], sc[1][j]), fmaxf(sc[2][j], sc[3][j]));
      for (int d = 8; d >= 1; d >>= 1) rm = fmaxf(rm, __shfl_xor(rm, d, 32));
      const float mn    = fmaxf(m[j], rm);
      const float alpha = __expf(m[j] - mn);
      float ps = 0.0f;
      for (int nt = 0; nt < 4; ++nt) {
        const float p = __expf(sc[nt][j] - mn);
        ps += p;
        lP[wave][(j + hsel * 8) * 64 + nt * 16 + l15] = f2bfu(p);
      }
      for (int d = 8; d >= 1; d >>= 1) ps += __shfl_xor(ps, d, 32);
      lsum[j] = lsum[j] * alpha + ps;
      m[j] = mn;
      for (int t = 0; t < 4; ++t) o[t][j] *= alpha;
    }

    // O += P * V  (P re-loaded from per-wave LDS in A-fragment layout)
    for (int s2 = 0; s2 < 2; ++s2) {
      Frag pa;
      pa.q[0] = *(const u32x4*)&lP[wave][l15 * 64 + s2 * 32 + hsel * 8];
      pa.q[1] = *(const u32x4*)&lP[wave][l15 * 64 + s2 * 32 + 16 + hsel * 8];
      Frag bf[4];                                  // B = V: n=hd, K-dim=key
      for (int nt2 = 0; nt2 < 4; ++nt2) {
        bf[nt2].q[0] = *(const u32x4*)&lVt[buf][(nt2 * 16 + l15) * 64 + s2 * 32 + hsel * 8];
        bf[nt2].q[1] = *(const u32x4*)&lVt[buf][(nt2 * 16 + l15) * 64 + s2 * 32 + 16 + hsel * 8];
      }
      SCHED_FENCE();
      for (int nt2 = 0; nt2 < 4; ++nt2)
        o[nt2] = wmma_bf16(pa, bf[nt2], o[nt2]);
      SCHED_FENCE();
    }

    if (more) attn_store_vt(lVt[buf ^ 1], vreg, tid);
    asm volatile("s_wait_asynccnt 0" ::: "memory");
    __syncthreads();
    buf ^= 1;
  }

  // Normalize and write attention output as f32 [B,S,D] (heads re-merged)
  for (int nt2 = 0; nt2 < 4; ++nt2) {
    for (int j = 0; j < 8; ++j) {
      const int sidx = q0 + j + hsel * 8;
      const int d    = h * HDD + nt2 * 16 + l15;
      const float rl = 1.0f / fmaxf(lsum[j], 1e-20f);
      out[(b * SS + sidx) * DD + d] = o[nt2][j] * rl;
    }
  }
}

// ---------------------------------------------------------------------------
extern "C" void kernel_launch(void* const* d_in, const int* in_sizes, int n_in,
                              void* d_out, int out_size, void* d_ws, size_t ws_size,
                              hipStream_t stream) {
  const float* x  = (const float*)d_in[0];
  const int* kpm  = (const int*)d_in[1];
  const float* Wq = (const float*)d_in[2];
  const float* bq = (const float*)d_in[3];
  const float* Wk = (const float*)d_in[4];
  const float* bk = (const float*)d_in[5];
  const float* Wv = (const float*)d_in[6];
  const float* bv = (const float*)d_in[7];
  const float* Wo = (const float*)d_in[8];
  const float* bo = (const float*)d_in[9];
  float* out = (float*)d_out;

  const size_t qkv_elems = (size_t)BB * HH * SS * HDD;   // bf16 elements
  unsigned short* qb = (unsigned short*)d_ws;
  unsigned short* kb = qb + qkv_elems;
  unsigned short* vb = kb + qkv_elems;
  float* attn = (float*)(vb + qkv_elems);                // f32 [B,S,D]

  dim3 gGemm(DD / 128, (BB * SS) / 128);   // (8, 32)
  mha_gemm_bf16wmma<<<gGemm, 256, 0, stream>>>(x, Wq, bq, qb, 1);
  mha_gemm_bf16wmma<<<gGemm, 256, 0, stream>>>(x, Wk, bk, kb, 1);
  mha_gemm_bf16wmma<<<gGemm, 256, 0, stream>>>(x, Wv, bv, vb, 1);

  dim3 gAttn(SS / 64, HH, BB);             // (32, 16, 2)
  mha_attn_flash<<<gAttn, 128, 0, stream>>>(qb, kb, vb, kpm, attn);

  mha_gemm_bf16wmma<<<gGemm, 256, 0, stream>>>(attn, Wo, bo, (void*)out, 0);
}
